// MoeSpaceTimeModel_89498528514777
// MI455X (gfx1250) — matrix operand order repaired and checked by reading, not compiled, not run on hardware
//
#include <hip/hip_runtime.h>

typedef __attribute__((ext_vector_type(16))) __bf16 v16bf;
typedef __attribute__((ext_vector_type(8)))  float  v8f;

#define WMMA_BF16(a, b, c) \
    __builtin_amdgcn_wmma_f32_16x16x32_bf16(false, (a), false, (b), (short)0, (c), false, false)

// ---------------- workspace fragment layout (bf16 elements) ----------------
// WpF : [8 experts][512]                  (B frag, K padded 16->32)
// W1F : [8][2 ktiles][4 ntiles][512]
// W2F : [8][2 ktiles][512]                (B frag, N padded 8->16)
#define WS_WP    0
#define WS_W1    4096
#define WS_W2    36864
#define WS_TOTAL 45056

// ---- fast device math: single-instruction transcendental paths ------------
__device__ __forceinline__ float frcp_(float x)  { return __builtin_amdgcn_rcpf(x); }
__device__ __forceinline__ float fsqrt_(float x) { return __builtin_amdgcn_sqrtf(x); }
__device__ __forceinline__ float fexp_(float x)  { return __expf(x); }      // v_exp_f32
__device__ __forceinline__ float fsin_(float x)  { return __sinf(x); }      // v_sin_f32
__device__ __forceinline__ float fcos_(float x)  { return __cosf(x); }      // v_cos_f32
__device__ __forceinline__ float silu_(float x)    { return x * frcp_(1.f + fexp_(-x)); }
__device__ __forceinline__ float sigmoid_(float x) { return frcp_(1.f + fexp_(-x)); }
__device__ __forceinline__ int   clampi_(int v, int lo, int hi) { return v < lo ? lo : (v > hi ? hi : v); }

// ---------------------------------------------------------------------------
// Parameter bundle (all f32 global pointers, insertion-order of setup_inputs)
// ---------------------------------------------------------------------------
struct P {
    const float *pos, *dir, *t;
    // gate_geo / gate_table / gate_render (Wp,bp,W1,b1,g,W2,b2)
    const float *gg_Wp, *gg_bp, *gg_W1, *gg_b1, *gg_g, *gg_W2, *gg_b2;
    const float *gt_Wp, *gt_bp, *gt_W1, *gt_b1, *gt_g, *gt_W2, *gt_b2;
    const float *gr_Wp, *gr_bp, *gr_W1, *gr_b1, *gr_g, *gr_W2, *gr_b2;
    // geo experts
    const float *geo_origin, *geo_rW1, *geo_rb1, *geo_rW2, *geo_rb2;
    const float *geo_fW1, *geo_fW2, *geo_fb2, *geo_frms, *geo_fW3, *geo_fb3;
    // tables [6][64][64][64][8][16]
    const float *tables;
    // render seg_glu f32 side params
    const float *r_bp, *r_b1, *r_g, *r_b2;
    // heads
    const float *alpha_W, *alpha_b, *color_W, *color_b;
    // prepped bf16 weight fragments
    const __bf16 *wsfrag;
    float *out;
    int N;
};

// ---------------------------------------------------------------------------
// prep: f32 render weights -> bf16 B-fragments in d_ws
// B layout (bf16, 32x16 per WMMA): lane l -> column (l&15), half = l>>4,
// element e -> K = e + 8*half (e<8), K = e + 8 + 8*half (e>=8) (within the 32-K tile).
// ---------------------------------------------------------------------------
__global__ void prep_frags(const float* __restrict__ Wp,   // [8][16][16]
                           const float* __restrict__ W1,   // [8][64][64]
                           const float* __restrict__ W2,   // [8][64][8]
                           __bf16* __restrict__ ws) {
    int gid = blockIdx.x * blockDim.x + threadIdx.x;
    if (gid >= WS_TOTAL) return;
    float v = 0.f;
    if (gid < WS_W1) {                           // WpF
        int e = gid >> 9, r = gid & 511;
        int lane = r >> 4, el = r & 15;
        int half = lane >> 4, col = lane & 15;
        int K = (el < 8) ? (el + 8 * half) : (el + 8 + 8 * half); // 0..31; rows >=16 are zero pad
        v = (K < 16) ? Wp[e * 256 + K * 16 + col] : 0.f;
    } else if (gid < WS_W2) {                    // W1F
        int idx = gid - WS_W1;
        int e = idx >> 12, rem = idx & 4095;
        int kt = rem >> 11, nt = (rem >> 9) & 3, r = rem & 511;
        int lane = r >> 4, el = r & 15;
        int half = lane >> 4, col = lane & 15;
        int K = kt * 32 + ((el < 8) ? (el + 8 * half) : (el + 8 + 8 * half));
        v = W1[e * 4096 + K * 64 + nt * 16 + col];
    } else {                                     // W2F
        int idx = gid - WS_W2;
        int e = idx >> 10, kt = (idx >> 9) & 1, r = idx & 511;
        int lane = r >> 4, el = r & 15;
        int half = lane >> 4, col = lane & 15;
        int K = kt * 32 + ((el < 8) ? (el + 8 * half) : (el + 8 + 8 * half));
        v = (col < 8) ? W2[e * 512 + K * 8 + col] : 0.f;  // cols 8..15 zero pad
    }
    ws[gid] = (__bf16)v;
}

// ---------------------------------------------------------------------------
// scalar helpers (per-lane, f32)
// ---------------------------------------------------------------------------
template <int ON>
__device__ void seg_glu_gate4(const float x[4],
                              const float* Wp, const float* bp,
                              const float* W1, const float* b1,
                              const float* g,  const float* W2, const float* b2,
                              float out[ON]) {
    float h[16];
#pragma unroll
    for (int q = 0; q < 4; ++q) {
        float xp = bp[q];
#pragma unroll
        for (int i = 0; i < 4; ++i) xp += x[i] * Wp[i * 4 + q];
        float s = fsin_(xp), c = fcos_(xp);
        h[q] = s; h[4 + q] = silu_(s); h[8 + q] = c; h[12 + q] = silu_(c);
    }
    float h1[16], ss = 0.f;
#pragma unroll
    for (int o = 0; o < 16; ++o) {
        float a = b1[o];
#pragma unroll
        for (int i = 0; i < 16; ++i) a += h[i] * W1[i * 16 + o];
        h1[o] = a; ss += a * a;
    }
    float rinv = frcp_(fsqrt_(ss * (1.f / 16.f)) + 1e-8f);
#pragma unroll
    for (int o = 0; o < 16; ++o) h1[o] = g[o] * h1[o] * rinv;
#pragma unroll
    for (int e = 0; e < ON; ++e) {
        float a = b2[e];
#pragma unroll
        for (int i = 0; i < 16; ++i) a += h1[i] * W2[i * ON + e];
        out[e] = a;
    }
}

__device__ void topk2(const float* x, int n, float w[2], int sel[2]) {
    int i1 = 0; float v1 = x[0];
    for (int j = 1; j < n; ++j) if (x[j] > v1) { v1 = x[j]; i1 = j; }
    int i2 = -1; float v2 = -3.4e38f;
    for (int j = 0; j < n; ++j) if (j != i1 && x[j] > v2) { v2 = x[j]; i2 = j; }
    float ee = fexp_(v2 - v1);           // <= 1, safe
    float dn = frcp_(1.f + ee);
    w[0] = dn; w[1] = ee * dn;
    sel[0] = i1; sel[1] = i2;
}

__device__ void geo_expert(int e, const float pz[3], float tt, const float x4[4],
                           const P& p, float out3[3]) {
    float dx = pz[0] - p.geo_origin[e * 3 + 0];
    float dy = pz[1] - p.geo_origin[e * 3 + 1];
    float dz = pz[2] - p.geo_origin[e * 3 + 2];
    float dist = fsqrt_(dx * dx + dy * dy + dz * dz);
    float alpha = atan2f(dy, dx);
    float ratio = fminf(1.f, fmaxf(-1.f, dz * frcp_(dist)));
    float beta = acosf(ratio);
    float sph[4] = {alpha, beta, dist, tt};
    float res[3];
    {
        float r14[14];
#pragma unroll
        for (int h = 0; h < 14; ++h) {
            float a = p.geo_rb1[e * 14 + h];
#pragma unroll
            for (int i = 0; i < 4; ++i) a += sph[i] * p.geo_rW1[(e * 4 + i) * 14 + h];
            r14[h] = a;
        }
#pragma unroll
        for (int o = 0; o < 3; ++o) {
            float a = p.geo_rb2[e * 3 + o];
#pragma unroll
            for (int h = 0; h < 14; ++h) a += r14[h] * p.geo_rW2[(e * 14 + h) * 3 + o];
            res[o] = a;
        }
    }
    float f14[14];
#pragma unroll
    for (int h = 0; h < 14; ++h) {
        float xg = 0.f, gt = 0.f;
#pragma unroll
        for (int i = 0; i < 4; ++i) {
            const float* row = p.geo_fW1 + (e * 4 + i) * 28;
            xg += x4[i] * row[h];
            gt += x4[i] * row[14 + h];
        }
        f14[h] = silu_(gt) * xg;
    }
    float f2[14], ss = 0.f;
#pragma unroll
    for (int o = 0; o < 14; ++o) {
        float a = p.geo_fb2[e * 14 + o];
#pragma unroll
        for (int h = 0; h < 14; ++h) a += f14[h] * p.geo_fW2[(e * 14 + h) * 14 + o];
        f2[o] = a; ss += a * a;
    }
    float rinv = frcp_(fsqrt_(ss * (1.f / 14.f)) + 1e-8f);
#pragma unroll
    for (int o = 0; o < 14; ++o) f2[o] = p.geo_frms[e * 14 + o] * f2[o] * rinv;
#pragma unroll
    for (int o = 0; o < 3; ++o) {
        float a = p.geo_fb3[e * 3 + o];
#pragma unroll
        for (int h = 0; h < 14; ++h) a += f2[h] * p.geo_fW3[(e * 14 + h) * 3 + o];
        out3[o] = sigmoid_(res[o] + a);
    }
}

// ---------------------------------------------------------------------------
// fused main kernel: 128 threads = 4 waves, 32 points/wave, 2 WMMA M-tiles/wave
// ---------------------------------------------------------------------------
__global__ void __launch_bounds__(128)
moe_spacetime_main(P p) {
    __shared__ __bf16 tfBF[4][32][16];   // table_feat, A-source
    __shared__ float  wrS [4][32][8];    // render routing weights
    __shared__ float  xpS [4][16][16];   // xp staging (D layout -> row major)
    __shared__ float  hF32[4][16][64];   // pre-norm hidden
    __shared__ __bf16 hBF [4][16][64];   // normalized hidden, A-source
    __shared__ float  outS[4][32][8];    // accumulated render output

    const int tid  = threadIdx.x;
    const int wave = tid >> 5;
    const int lane = tid & 31;
    const int ptRaw = blockIdx.x * 128 + tid;
    const int pt = ptRaw < p.N ? ptRaw : (p.N - 1);

    // ================= phase 1: per-point scalar pipeline =================
    float pos3[3] = {p.pos[pt * 3], p.pos[pt * 3 + 1], p.pos[pt * 3 + 2]};
    float tval = p.t[pt];
    float x4[4] = {pos3[0], pos3[1], pos3[2], tval};

    // geo MoE: default expert 0 + top-1 routed (indices >= 7 dropped)
    float lg[8];
    seg_glu_gate4<8>(x4, p.gg_Wp, p.gg_bp, p.gg_W1, p.gg_b1, p.gg_g, p.gg_W2, p.gg_b2, lg);
    int sel = 0; float best = lg[0];
    for (int j = 1; j < 8; ++j) if (lg[j] > best) { best = lg[j]; sel = j; }
    float geo3[3];
    geo_expert(0, pos3, tval, x4, p, geo3);
    if (sel < 7) {
        float g2[3];
        geo_expert(1 + sel, pos3, tval, x4, p, g2);
        geo3[0] += g2[0]; geo3[1] += g2[1]; geo3[2] += g2[2];
    }

    float coords[4] = {geo3[0], geo3[1], geo3[2], tval};
    int ix = clampi_((int)(coords[0] * 63.f), 0, 63);
    int iy = clampi_((int)(coords[1] * 63.f), 0, 63);
    int iz = clampi_((int)(coords[2] * 63.f), 0, 63);
    int it = clampi_((int)(tval * 7.f), 0, 7);

    // table MoE: 2 defaults + top-2 of 6 (routed idx >= 4 dropped)
    float lt[6];
    seg_glu_gate4<6>(coords, p.gt_Wp, p.gt_bp, p.gt_W1, p.gt_b1, p.gt_g, p.gt_W2, p.gt_b2, lt);
    float wtk[2]; int stk[2];
    topk2(lt, 6, wtk, stk);
    float wtab[6] = {1.f, 1.f, 0.f, 0.f, 0.f, 0.f};
    if (stk[0] < 4) wtab[2 + stk[0]] += wtk[0];
    if (stk[1] < 4) wtab[2 + stk[1]] += wtk[1];

    float tf[16];
#pragma unroll
    for (int f = 0; f < 16; ++f) tf[f] = 0.f;
    for (int e = 0; e < 6; ++e) {
        float w = wtab[e];
        if (w != 0.f) {
            const float* src = p.tables +
                (((((size_t)e * 64 + ix) * 64 + iy) * 64 + iz) * 8 + it) * 16;
#pragma unroll
            for (int f = 0; f < 16; ++f) tf[f] += w * src[f];
        }
    }

    // render routing: 1 default + top-2 of 8 (routed idx >= 7 dropped)
    float lr[8];
    seg_glu_gate4<8>(coords, p.gr_Wp, p.gr_bp, p.gr_W1, p.gr_b1, p.gr_g, p.gr_W2, p.gr_b2, lr);
    float wrk[2]; int srk[2];
    topk2(lr, 8, wrk, srk);
    float wr[8] = {1.f, 0.f, 0.f, 0.f, 0.f, 0.f, 0.f, 0.f};
    if (srk[0] < 7) wr[1 + srk[0]] += wrk[0];
    if (srk[1] < 7) wr[1 + srk[1]] += wrk[1];

#pragma unroll
    for (int f = 0; f < 16; ++f) tfBF[wave][lane][f] = (__bf16)tf[f];
#pragma unroll
    for (int e = 0; e < 8; ++e) wrS[wave][lane][e] = wr[e];
    __syncthreads();

    // ================= phase 2: dense 8-expert render MoE via WMMA =========
    // A layout (bf16 16x32): lane -> row (lane&15); K = el + 8*half (el<8),
    //                        K = el + 8 + 8*half (el>=8)
    const int m    = lane & 15;
    const int half = lane >> 4;
    const int ncol = lane & 15;

    for (int tt = 0; tt < 2; ++tt) {
        v16bf a_tf;
#pragma unroll
        for (int el = 0; el < 8; ++el)
            a_tf[el] = tfBF[wave][tt * 16 + m][el + 8 * half];
#pragma unroll
        for (int el = 8; el < 16; ++el) a_tf[el] = (__bf16)0.0f;  // K 16..31 pad

        float ctot[8];
#pragma unroll
        for (int r = 0; r < 8; ++r) ctot[r] = 0.f;

        for (int e = 0; e < 8; ++e) {
            // --- xp = tf @ Wp + bp  (one WMMA, K padded to 32) ---
            v16bf bwp = *reinterpret_cast<const v16bf*>(p.wsfrag + WS_WP + (size_t)e * 512 + lane * 16);
            v8f xpd = {};
            xpd = WMMA_BF16(a_tf, bwp, xpd);
            float bpv = p.r_bp[e * 16 + ncol];
#pragma unroll
            for (int r = 0; r < 8; ++r)
                xpS[wave][r + 8 * half][ncol] = xpd[r] + bpv;     // D: M=r+8*half, N=ncol
            __syncthreads();

            // --- h = [sin, silu(sin), cos, silu(cos)] A-fragments (K=64) ---
            v16bf ah0, ah1;
#pragma unroll
            for (int el = 0; el < 16; ++el) {
                int Ka = (el < 8) ? (el + 8 * half) : (el + 8 + 8 * half); // 0..31
                float xv = xpS[wave][m][Ka & 15];
                float sv = fsin_(xv), cv = fcos_(xv);
                ah0[el] = (__bf16)((Ka < 16) ? sv : silu_(sv));   // features 0..31
                ah1[el] = (__bf16)((Ka < 16) ? cv : silu_(cv));   // features 32..63
            }

            // --- h @ W1 + b1  (4 N-tiles x 2 K-tiles = 8 WMMAs) ---
            const __bf16* w1base = p.wsfrag + WS_W1 + (size_t)e * 4096;
#pragma unroll
            for (int nt = 0; nt < 4; ++nt) {
                v16bf bk0 = *reinterpret_cast<const v16bf*>(w1base + nt * 512 + lane * 16);
                v16bf bk1 = *reinterpret_cast<const v16bf*>(w1base + 2048 + nt * 512 + lane * 16);
                v8f acc = {};
                acc = WMMA_BF16(ah0, bk0, acc);
                acc = WMMA_BF16(ah1, bk1, acc);
                float b1v = p.r_b1[e * 64 + nt * 16 + ncol];
#pragma unroll
                for (int r = 0; r < 8; ++r)
                    hF32[wave][r + 8 * half][nt * 16 + ncol] = acc[r] + b1v;
            }
            __syncthreads();

            // --- rmsnorm rows (d=64), write bf16 ---
            if (lane < 16) {
                float ss = 0.f;
                for (int j = 0; j < 64; ++j) { float v = hF32[wave][lane][j]; ss += v * v; }
                float rinv = frcp_(fsqrt_(ss) * 0.125f + 1e-8f);
                for (int j = 0; j < 64; ++j)
                    hBF[wave][lane][j] = (__bf16)(p.r_g[e * 64 + j] * hF32[wave][lane][j] * rinv);
            }
            __syncthreads();

            // --- hn @ W2 + b2  (2 WMMAs, N padded 8->16) ---
            v16bf an0, an1;
#pragma unroll
            for (int el = 0; el < 16; ++el) {
                int Ka = (el < 8) ? (el + 8 * half) : (el + 8 + 8 * half);
                an0[el] = hBF[wave][m][Ka];
                an1[el] = hBF[wave][m][32 + Ka];
            }
            const __bf16* w2base = p.wsfrag + WS_W2 + (size_t)e * 1024;
            v16bf c0 = *reinterpret_cast<const v16bf*>(w2base + lane * 16);
            v16bf c1 = *reinterpret_cast<const v16bf*>(w2base + 512 + lane * 16);
            v8f a2 = {};
            a2 = WMMA_BF16(an0, c0, a2);
            a2 = WMMA_BF16(an1, c1, a2);
            float b2v = (ncol < 8) ? p.r_b2[e * 8 + ncol] : 0.f;
#pragma unroll
            for (int r = 0; r < 8; ++r) {
                float w = wrS[wave][tt * 16 + r + 8 * half][e];
                ctot[r] += w * (a2[r] + b2v);
            }
        }

        if (ncol < 8) {
#pragma unroll
            for (int r = 0; r < 8; ++r)
                outS[wave][tt * 16 + r + 8 * half][ncol] = ctot[r];
        }
    }
    __syncthreads();

    // ================= phase 3: heads, per point ==========================
    float rend[8];
#pragma unroll
    for (int f = 0; f < 8; ++f) rend[f] = outS[wave][lane][f];

    float opac = p.alpha_b[0];
#pragma unroll
    for (int f = 0; f < 8; ++f) opac += rend[f] * p.alpha_W[f];

    float cin[12];
#pragma unroll
    for (int f = 0; f < 8; ++f) cin[f] = rend[f];
    cin[8]  = p.dir[pt * 3 + 0];
    cin[9]  = p.dir[pt * 3 + 1];
    cin[10] = p.dir[pt * 3 + 2];
    cin[11] = opac;

    if (ptRaw < p.N) {
#pragma unroll
        for (int c = 0; c < 3; ++c) {
            float a = p.color_b[c];
#pragma unroll
            for (int i = 0; i < 12; ++i) a += cin[i] * p.color_W[i * 3 + c];
            p.out[(size_t)pt * 4 + c] = a;
        }
        p.out[(size_t)pt * 4 + 3] = opac;
    }
}

// ---------------------------------------------------------------------------
extern "C" void kernel_launch(void* const* d_in, const int* in_sizes, int n_in,
                              void* d_out, int out_size, void* d_ws, size_t ws_size,
                              hipStream_t stream) {
    P p;
    p.pos = (const float*)d_in[0];
    p.dir = (const float*)d_in[1];
    p.t   = (const float*)d_in[2];
    // params flattened in insertion order:
    // gate_geo (3..9), gate_table (10..16), gate_render (17..23)
    p.gg_Wp = (const float*)d_in[3];  p.gg_bp = (const float*)d_in[4];
    p.gg_W1 = (const float*)d_in[5];  p.gg_b1 = (const float*)d_in[6];
    p.gg_g  = (const float*)d_in[7];  p.gg_W2 = (const float*)d_in[8];
    p.gg_b2 = (const float*)d_in[9];
    p.gt_Wp = (const float*)d_in[10]; p.gt_bp = (const float*)d_in[11];
    p.gt_W1 = (const float*)d_in[12]; p.gt_b1 = (const float*)d_in[13];
    p.gt_g  = (const float*)d_in[14]; p.gt_W2 = (const float*)d_in[15];
    p.gt_b2 = (const float*)d_in[16];
    p.gr_Wp = (const float*)d_in[17]; p.gr_bp = (const float*)d_in[18];
    p.gr_W1 = (const float*)d_in[19]; p.gr_b1 = (const float*)d_in[20];
    p.gr_g  = (const float*)d_in[21]; p.gr_W2 = (const float*)d_in[22];
    p.gr_b2 = (const float*)d_in[23];
    // geo (24..34)
    p.geo_origin = (const float*)d_in[24];
    p.geo_rW1 = (const float*)d_in[25]; p.geo_rb1 = (const float*)d_in[26];
    p.geo_rW2 = (const float*)d_in[27]; p.geo_rb2 = (const float*)d_in[28];
    p.geo_fW1 = (const float*)d_in[29]; p.geo_fW2 = (const float*)d_in[30];
    p.geo_fb2 = (const float*)d_in[31]; p.geo_frms = (const float*)d_in[32];
    p.geo_fW3 = (const float*)d_in[33]; p.geo_fb3 = (const float*)d_in[34];
    // tables (35)
    p.tables = (const float*)d_in[35];
    // render (36..42): Wp,bp,W1,b1,g,W2,b2
    const float* r_Wp = (const float*)d_in[36];
    p.r_bp = (const float*)d_in[37];
    const float* r_W1 = (const float*)d_in[38];
    p.r_b1 = (const float*)d_in[39];
    p.r_g  = (const float*)d_in[40];
    const float* r_W2 = (const float*)d_in[41];
    p.r_b2 = (const float*)d_in[42];
    // heads (43..46)
    p.alpha_W = (const float*)d_in[43];
    p.alpha_b = (const float*)d_in[44];
    p.color_W = (const float*)d_in[45];
    p.color_b = (const float*)d_in[46];

    p.wsfrag = reinterpret_cast<const __bf16*>(d_ws);
    p.out = (float*)d_out;
    p.N = in_sizes[2];   // t has N elements

    // 1) build bf16 WMMA B-fragments of render weights in d_ws (90 KB)
    prep_frags<<<(WS_TOTAL + 255) / 256, 256, 0, stream>>>(
        r_Wp, r_W1, r_W2, reinterpret_cast<__bf16*>(d_ws));

    // 2) fused forward: 128 threads/block, 128 points/block
    int blocks = (p.N + 127) / 128;
    moe_spacetime_main<<<blocks, 128, 0, stream>>>(p);
}
